// MyLSTM_17678085390655
// MI455X (gfx1250) — matrix operand order repaired and checked by reading, not compiled, not run on hardware
//
#include <hip/hip_runtime.h>
#include <hip/hip_bf16.h>

// Problem dims (fixed by the reference)
#define BATCH 32
#define SEQ   1024
#define DDIM  1024
#define UNITS 1024
#define NGATE 4
#define KCOL  (NGATE * UNITS)   // 4096

// Workspace layout (bytes). Total ~82 MB.
#define OFF_SYNC 0ull
#define OFF_H    4096ull                     // 2 * 32*1024 bf16 = 131072 B (double buffer)
#define OFF_WB   262144ull                   // kernel bf16: 1024*4096*2 = 8 MB
#define OFF_RB   9437184ull                  // recurrent bf16: 8 MB
#define OFF_XB   18874368ull                 // x bf16: 32*1024*1024*2 = 64 MB

typedef __attribute__((ext_vector_type(16))) __bf16 v16bf;
typedef __attribute__((ext_vector_type(8)))  __bf16 v8bf;
typedef __attribute__((ext_vector_type(8)))  float  v8f;

__device__ __forceinline__ unsigned short f2bf(float f) {
  unsigned u = __float_as_uint(f);
  u += 0x7FFFu + ((u >> 16) & 1u);           // round-to-nearest-even
  return (unsigned short)(u >> 16);
}

__device__ __forceinline__ float hsig(float z) {
  return fminf(fmaxf(0.2f * z + 0.5f, 0.0f), 1.0f);
}

__device__ __forceinline__ v16bf cat8(v8bf lo, v8bf hi) {
  return __builtin_shufflevector(lo, hi, 0,1,2,3,4,5,6,7,8,9,10,11,12,13,14,15);
}

// fp32 -> bf16 pack, 4 elements/thread
__global__ void cvt_f32_bf16(const float* __restrict__ src,
                             unsigned short* __restrict__ dst, int n4) {
  int i = blockIdx.x * blockDim.x + threadIdx.x;
  if (i < n4) {
    float4 v = ((const float4*)src)[i];
    ushort4 o;
    o.x = f2bf(v.x); o.y = f2bf(v.y); o.z = f2bf(v.z); o.w = f2bf(v.w);
    ((ushort4*)dst)[i] = o;
  }
}

// Device-wide barrier (arrival counter + generation), all blocks co-resident.
__device__ __forceinline__ void gridbar(unsigned* cnt, unsigned* gen, unsigned nblk) {
  __syncthreads();
  __threadfence();
  if (threadIdx.x == 0) {
    unsigned g = __hip_atomic_load(gen, __ATOMIC_RELAXED, __HIP_MEMORY_SCOPE_AGENT);
    unsigned prev = __hip_atomic_fetch_add(cnt, 1u, __ATOMIC_ACQ_REL, __HIP_MEMORY_SCOPE_AGENT);
    if (prev == nblk - 1u) {
      __hip_atomic_store(cnt, 0u, __ATOMIC_RELAXED, __HIP_MEMORY_SCOPE_AGENT);
      __hip_atomic_fetch_add(gen, 1u, __ATOMIC_RELEASE, __HIP_MEMORY_SCOPE_AGENT);
    } else {
      while (__hip_atomic_load(gen, __ATOMIC_ACQUIRE, __HIP_MEMORY_SCOPE_AGENT) == g) {
        __builtin_amdgcn_s_sleep(8);
      }
    }
  }
  __syncthreads();
}

// Persistent fused LSTM scan: z = [x_s ; h] @ [W ; R] per step, gates, grid barrier.
// Grid: 64 blocks (one per 16-column u-tile) x 256 threads (8 waves).
// Wave w: gate g = w&3 (0=i,1=f,2=c,3=o), half = w>>2 (0: x@W, 1: h@R).
// Each wave computes both 16-row M-tiles (B-tile register reuse x2).
__launch_bounds__(256)
__global__ void lstm_scan(const unsigned short* __restrict__ xb,
                          const unsigned short* __restrict__ wb,
                          const unsigned short* __restrict__ rb,
                          unsigned short* __restrict__ h0,
                          unsigned short* __restrict__ h1,
                          float* __restrict__ out,
                          unsigned* __restrict__ sync) {
  __shared__ float zbuf[8][2][256];   // [wave][mtile][16x16 fp32 z-partial] = 16 KB

  const int tid   = threadIdx.x;
  const int w     = tid >> 5;
  const int lane  = tid & 31;
  const int g     = w & 3;
  const int half  = w >> 2;
  const int u0    = blockIdx.x * 16;
  const int colbase = g * UNITS + u0;

  // A-matrix lane mapping (16-bit A, 16x32): lanes 0-15 rows 0-15 / K lo-half,
  // lanes 16-31 same rows / K hi-half; elements 0..7 at k+khalf, 8..15 at k+16+khalf.
  const int arow  = lane & 15;
  const int khalf = (lane >> 4) * 8;

  const unsigned short* bmat = half ? rb : wb;

  // Per-thread carried cell state (this (b,u) partition is private to this thread).
  float c0 = 0.0f, c1 = 0.0f;

  for (int s = 0; s < SEQ; ++s) {
    const unsigned short* hcur = (s & 1) ? h1 : h0;
    unsigned short*       hnxt = (s & 1) ? h0 : h1;

    v8f acc0 = {};
    v8f acc1 = {};

    #pragma unroll 4
    for (int kk = 0; kk < 32; ++kk) {
      const int k0 = kk * 32;
      // B tile (32x16 bf16): lane = N column (lane&15), elements 0..7 at k+khalf, 8..15 at k+16+khalf.
      const unsigned short* bp = bmat + (size_t)(k0 + khalf) * KCOL + colbase + arow;
      v16bf B;
      #pragma unroll
      for (int i = 0; i < 8; ++i) {
        B[i]     = ((const __bf16*)bp)[(size_t)i * KCOL];
        B[8 + i] = ((const __bf16*)bp)[(size_t)(16 + i) * KCOL];
      }
      __builtin_prefetch(bp + 32 * KCOL, 0, 0);   // next K-chunk -> global_prefetch_b8

      v16bf A0, A1;
      if (half == 0) {
        // A = x[:, s, :] slice, layout (B,S,D)
        const unsigned short* a0p = xb + ((size_t)( arow       * SEQ + s)) * DDIM + k0 + khalf;
        const unsigned short* a1p = xb + ((size_t)((arow + 16) * SEQ + s)) * DDIM + k0 + khalf;
        A0 = cat8(*(const v8bf*)a0p, *(const v8bf*)(a0p + 16));
        A1 = cat8(*(const v8bf*)a1p, *(const v8bf*)(a1p + 16));
      } else {
        // A = h (32 x 1024 bf16, double-buffered)
        const unsigned short* a0p = hcur + (size_t)( arow      ) * UNITS + k0 + khalf;
        const unsigned short* a1p = hcur + (size_t)(arow + 16) * UNITS + k0 + khalf;
        A0 = cat8(*(const v8bf*)a0p, *(const v8bf*)(a0p + 16));
        A1 = cat8(*(const v8bf*)a1p, *(const v8bf*)(a1p + 16));
      }

      acc0 = __builtin_amdgcn_wmma_f32_16x16x32_bf16(false, A0, false, B, (short)0, acc0, false, false);
      acc1 = __builtin_amdgcn_wmma_f32_16x16x32_bf16(false, A1, false, B, (short)0, acc1, false, false);
    }

    // C/D layout: VGPR j, lane l -> (M = j + 8*(l>>4), N = l&15)
    {
      const int n  = lane & 15;
      const int mb = (lane >> 4) * 8;
      #pragma unroll
      for (int j = 0; j < 8; ++j) {
        zbuf[w][0][(mb + j) * 16 + n] = acc0[j];
        zbuf[w][1][(mb + j) * 16 + n] = acc1[j];
      }
    }
    __syncthreads();

    // Gate combine: 512 (b,u) elements per block, 2 per thread; c stays in regs.
    #pragma unroll
    for (int e2 = 0; e2 < 2; ++e2) {
      const int e   = tid + e2 * 256;
      const int mt  = e >> 8;
      const int idx = e & 255;
      float zi = zbuf[0][mt][idx] + zbuf[4][mt][idx];
      float zf = zbuf[1][mt][idx] + zbuf[5][mt][idx];
      float zc = zbuf[2][mt][idx] + zbuf[6][mt][idx];
      float zo = zbuf[3][mt][idx] + zbuf[7][mt][idx];

      float& c = e2 ? c1 : c0;
      float ft = hsig(zf);
      float it = hsig(zi);
      float cb = tanhf(zc);
      c = tanhf(ft * c + it * cb);          // reference carries tanh(c_new)
      float h = hsig(zo) * c;

      const int r = idx >> 4, n = idx & 15;
      const int b = mt * 16 + r;
      const int u = u0 + n;
      out[((size_t)b * SEQ + s) * UNITS + u] = h;          // (B,S,U) fp32
      hnxt[b * UNITS + u] = f2bf(h);
    }

    gridbar(sync, sync + 1, gridDim.x);
  }
}

extern "C" void kernel_launch(void* const* d_in, const int* in_sizes, int n_in,
                              void* d_out, int out_size, void* d_ws, size_t ws_size,
                              hipStream_t stream) {
  const float* x = (const float*)d_in[0];
  const float* W = (const float*)d_in[1];
  const float* R = (const float*)d_in[2];
  float* out = (float*)d_out;
  char* ws = (char*)d_ws;

  unsigned*       sync = (unsigned*)(ws + OFF_SYNC);
  unsigned short* h0   = (unsigned short*)(ws + OFF_H);
  unsigned short* h1   = h0 + BATCH * UNITS;
  unsigned short* wb   = (unsigned short*)(ws + OFF_WB);
  unsigned short* rb   = (unsigned short*)(ws + OFF_RB);
  unsigned short* xb   = (unsigned short*)(ws + OFF_XB);

  // Zero barrier state + h double buffer (graph-capturable async memset).
  hipMemsetAsync(ws, 0, OFF_H + (size_t)2 * BATCH * UNITS * sizeof(unsigned short), stream);

  // fp32 -> bf16 packs (one pass each; W/R then live in L2 for all 1024 steps).
  const int nx4 = BATCH * SEQ * DDIM / 4;       // 8,388,608
  cvt_f32_bf16<<<nx4 / 256, 256, 0, stream>>>(x, xb, nx4);
  const int nw4 = DDIM * KCOL / 4;              // 1,048,576
  cvt_f32_bf16<<<nw4 / 256, 256, 0, stream>>>(W, wb, nw4);
  cvt_f32_bf16<<<nw4 / 256, 256, 0, stream>>>(R, rb, nw4);

  // Persistent fused scan: 64 co-resident blocks, device-wide barrier per step.
  lstm_scan<<<64, 256, 0, stream>>>(xb, wb, rb, h0, h1, out, sync);
}
